// CSA_ConvBlock_5136780886739
// MI455X (gfx1250) — matrix operand order repaired and checked by reading, not compiled, not run on hardware
//
#include <hip/hip_runtime.h>
#include <hip/hip_bf16.h>

typedef __attribute__((ext_vector_type(16))) __bf16          v16bf;
typedef __attribute__((ext_vector_type(8)))  float           v8f;
typedef __attribute__((ext_vector_type(8)))  unsigned short  u16x8;
typedef __attribute__((ext_vector_type(16))) unsigned short  u16x16;

__device__ __forceinline__ unsigned short f2bf(float f) {
    __bf16 b = (__bf16)f;
    return __builtin_bit_cast(unsigned short, b);
}

// ---------------------------------------------------------------------------
// K0x: convert x (fp32, 16*256*64*64) -> bf16 copy in ws (4 elems/thread)
// ---------------------------------------------------------------------------
__global__ void k0x_cvt(const float* __restrict__ x, unsigned short* __restrict__ xbf) {
    int i = blockIdx.x * blockDim.x + threadIdx.x;
    int base = i * 4;
    if (base >= 16 * 256 * 64 * 64) return;
#pragma unroll
    for (int j = 0; j < 4; ++j) xbf[base + j] = f2bf(x[base + j]);
}

// ---------------------------------------------------------------------------
// K0: pack wv (fp32 [256][256][3][3]) -> bf16 A fragments, laid out so each
// half-wave's 16 fragment values are one contiguous 32B chunk:
//   Apk[((m*72 + chunk)*2 + lhalf)*16 + j]
// where chunk = (kh*3+kw)*8 + ci/32, and element j maps to K-offset
//   lhalf*16 + j                (K 0..15 / 16..31 of the 32-chunk)
// matching the contiguous-K B fragments from LDS.  K' = (kh*3+kw)*256 + ci.
// ---------------------------------------------------------------------------
__global__ void k0_packw(const float* __restrict__ wv, unsigned short* __restrict__ Apk) {
    int i = blockIdx.x * blockDim.x + threadIdx.x;
    if (i >= 256 * 2304) return;
    int j  = i & 15;
    int lh = (i >> 4) & 1;
    int q  = (i >> 5) % 72;
    int m  = i / 2304;
    int off = lh * 16 + j;
    int kprime = q * 32 + off;
    int khkw = kprime >> 8, ci = kprime & 255;
    Apk[i] = f2bf(wv[(size_t)m * 2304 + ci * 9 + khkw]);
}

// ---------------------------------------------------------------------------
// K1: column sums of x with edge variants.
//     CS[b][ci][v][w]: v=0 sum h=0..62 ; v=1 sum h=0..63 ; v=2 sum h=1..63
// ---------------------------------------------------------------------------
__global__ void k1_colsum(const float* __restrict__ x, float* __restrict__ CS) {
    int idx = blockIdx.x * blockDim.x + threadIdx.x;  // B*C*W = 262144
    if (idx >= 16 * 256 * 64) return;
    int w = idx & 63;
    int bc = idx >> 6;  // b*256+ci
    const float* col = x + (size_t)bc * 4096 + w;
    float s = 0.f, x0 = 0.f, xl = 0.f;
    for (int h = 0; h < 64; ++h) {
        float v = col[h * 64];
        s += v;
        if (h == 0) x0 = v;
        if (h == 63) xl = v;
    }
    float* o = CS + (size_t)bc * 192 + w;
    o[0]   = s - xl;  // kh=0 variant (input rows -1..62, pad row dropped)
    o[64]  = s;       // kh=1 variant
    o[128] = s - x0;  // kh=2 variant
}

// ---------------------------------------------------------------------------
// K2a: wksum[ci*9+kh*3+kw] = sum_c wk[c][ci][kh][kw]
// ---------------------------------------------------------------------------
__global__ void k2a_wksum(const float* __restrict__ wk, float* __restrict__ wks) {
    int j = blockIdx.x * blockDim.x + threadIdx.x;
    if (j >= 2304) return;
    float a = 0.f;
    for (int c = 0; c < 256; ++c) a += wk[(size_t)c * 2304 + j];
    wks[j] = a;
}

// ---------------------------------------------------------------------------
// K2b: k_sum[b][w] = sum_{ci,kh,kw} wksum * CS[b,ci,kh, w+kw-1]
// ---------------------------------------------------------------------------
__global__ void k2b_ksum(const float* __restrict__ CS, const float* __restrict__ wks,
                         float* __restrict__ ksum) {
    int idx = blockIdx.x * blockDim.x + threadIdx.x;  // 16*64
    if (idx >= 16 * 64) return;
    int b = idx >> 6, w = idx & 63;
    float acc = 0.f;
    for (int ci = 0; ci < 256; ++ci) {
        const float* csb = CS + (size_t)(b * 256 + ci) * 192;
        const float* wk9 = wks + ci * 9;
        for (int kh = 0; kh < 3; ++kh)
            for (int kw = 0; kw < 3; ++kw) {
                int sw = w + kw - 1;
                if (sw >= 0 && sw < 64) acc += wk9[kh * 3 + kw] * csb[kh * 64 + sw];
            }
    }
    ksum[idx] = acc;
}

// ---------------------------------------------------------------------------
// K3a: T[b][ci*9+kh*3+kw] = sum_w CS[b,ci,kh, w+kw-1] * k_sum[b,w]
// ---------------------------------------------------------------------------
__global__ void k3a_T(const float* __restrict__ CS, const float* __restrict__ ksum,
                      float* __restrict__ T) {
    int idx = blockIdx.x * blockDim.x + threadIdx.x;  // 16*2304
    if (idx >= 16 * 2304) return;
    int b = idx / 2304, j = idx % 2304;
    int ci = j / 9, r = j % 9, kh = r / 3, kw = r % 3;
    const float* cs = CS + ((size_t)(b * 256 + ci) * 3 + kh) * 64;
    const float* ks = ksum + b * 64;
    float acc = 0.f;
    for (int w = 0; w < 64; ++w) {
        int sw = w + kw - 1;
        if (sw >= 0 && sw < 64) acc += cs[sw] * ks[w];
    }
    T[idx] = acc;
}

// ---------------------------------------------------------------------------
// K3b: f_scores[b,c] = (wq[c,:] . T[b,:]) / (sqrt(64)*64*64) ; softmax over c
// ---------------------------------------------------------------------------
__global__ void __launch_bounds__(256) k3b_scores(const float* __restrict__ wq,
                                                  const float* __restrict__ T,
                                                  float* __restrict__ scores) {
    __shared__ float tsh[2304];
    __shared__ float red[256];
    int b = blockIdx.x, c = threadIdx.x;
    for (int j = c; j < 2304; j += 256) tsh[j] = T[b * 2304 + j];
    __syncthreads();
    const float* wrow = wq + (size_t)c * 2304;
    float acc = 0.f;
    for (int j = 0; j < 2304; ++j) acc += wrow[j] * tsh[j];
    acc *= (1.0f / 32768.0f);  // 1/(sqrt(W)*H*H) = 1/(8*4096)
    red[c] = acc;
    __syncthreads();
    for (int s = 128; s > 0; s >>= 1) {
        if (c < s) red[c] = fmaxf(red[c], red[c + s]);
        __syncthreads();
    }
    float m = red[0];
    __syncthreads();
    float e = __expf(acc - m);
    red[c] = e;
    __syncthreads();
    for (int s = 128; s > 0; s >>= 1) {
        if (c < s) red[c] += red[c + s];
        __syncthreads();
    }
    scores[b * 256 + c] = e / red[0];
}

// ---------------------------------------------------------------------------
// K4: implicit-GEMM conv3x3(x, wv) via bf16 WMMA, fused epilogue:
//     out = relu( (score[b,c]*fv + x - mean[c]) * gamma[c]/sqrt(var[c]+eps) + beta[c] )
//
// One block = 256 thr (8 waves) computes M=256 (all C_out) x N=64 (one (b,h)
// row).  Each wave owns M rows {wave*16, wave*16+128} so every B fragment
// feeds 2 WMMAs.  LDS x-tile is [N=64][K=32] bf16 with column stride padded
// to 40 u16 (80 B): 16B-aligned b128 frag reads, bank-conflict-free.
// A fragments are single 32B contiguous loads from the pre-packed layout.
// Double-buffered: tile t+1 global load issues before WMMA work on tile t.
// One staged tile is reused by all three kw shifts.
// ---------------------------------------------------------------------------
#define LDS_CSTRIDE 40  // u16 units per column (padded from 32)

__global__ void __launch_bounds__(256) k4_convv_wmma(
    const unsigned short* __restrict__ xbf, const float* __restrict__ x,
    const unsigned short* __restrict__ Apk, const float* __restrict__ scores,
    const float* __restrict__ gamma, const float* __restrict__ beta,
    const float* __restrict__ rmean, const float* __restrict__ rvar,
    float* __restrict__ out) {
    __shared__ __attribute__((aligned(64))) unsigned short ldsB[2][64 * LDS_CSTRIDE];

    const int bh = blockIdx.x;
    const int b = bh >> 6;
    const int h = bh & 63;
    const int tid = (int)threadIdx.x;
    const int wave = tid >> 5;
    const int lane = tid & 31;
    const int lrow = lane & 15;
    const int lhalf = lane >> 4;
    const int m_base = wave * 16;

    const int kfill = tid >> 3;        // 0..31: ci offset this thread stages
    const int n0 = (tid & 7) * 8;      // 0,8,..,56: first column staged

    v8f acc0[4], acc1[4];
#pragma unroll
    for (int g = 0; g < 4; ++g)
#pragma unroll
        for (int i = 0; i < 8; ++i) { acc0[g][i] = 0.f; acc1[g][i] = 0.f; }

    const unsigned short* xb = xbf + (size_t)b * 256 * 4096;
    const u16x16 zvec = (u16x16)0;

    // tile index t = kh*8 + cc  (kh in 0..2, cc = ci chunk in 0..7)
    auto tile_src = [&](int t) -> const unsigned short* {
        const int kh = t >> 3, cc = t & 7;
        const int hr = h + kh - 1;
        if (hr < 0 || hr >= 64) return nullptr;
        return xb + (size_t)(cc * 32 + kfill) * 4096 + hr * 64 + n0;
    };
    // A fragment: one contiguous 32B chunk per half-wave from packed layout
    auto load_a = [&](int row, int chunk) -> v16bf {
        const unsigned short* ap =
            Apk + ((((size_t)row * 72 + chunk) * 2 + lhalf) << 4);
        u16x16 aa = *(const u16x16*)ap;  // 2x global_load_b128, no shuffle
        return __builtin_bit_cast(v16bf, aa);
    };

    // ---- preload + stage tile 0 ----
    {
        u16x8 v = (u16x8)0;
        const unsigned short* p = tile_src(0);
        if (p) v = *(const u16x8*)p;  // global_load_b128
#pragma unroll
        for (int j = 0; j < 8; ++j) ldsB[0][(n0 + j) * LDS_CSTRIDE + kfill] = v[j];
    }
    __syncthreads();

    for (int t = 0; t < 24; ++t) {
        const int buf = t & 1;
        // ---- issue global load for tile t+1 (overlaps WMMA below) ----
        u16x8 vnext = (u16x8)0;
        if (t < 23) {
            const unsigned short* p = tile_src(t + 1);
            if (p) vnext = *(const u16x8*)p;
            if (t < 22) {
                const unsigned short* q = tile_src(t + 2);
                if (q) __builtin_prefetch(q, 0, 0);  // global_prefetch_b8
            }
        }
        // ---- WMMA on tile t: 3 kw shifts x 4 N-groups x 2 M-halves ----
        const int kh = t >> 3, cc = t & 7;
#pragma unroll
        for (int kw = 0; kw < 3; ++kw) {
            const int chunk = (kh * 3 + kw) * 8 + cc;  // K-chunk index (32 wide)
            v16bf af0 = load_a(m_base, chunk);
            v16bf af1 = load_a(m_base + 128, chunk);
#pragma unroll
            for (int g = 0; g < 4; ++g) {
                const int cn = g * 16 + lrow + kw - 1;  // shifted source column
                u16x16 bb = (cn >= 0 && cn < 64)
                                ? *(const u16x16*)&ldsB[buf][cn * LDS_CSTRIDE + lhalf * 16]
                                : zvec;                  // 2x ds_load_b128
                v16bf bf = __builtin_bit_cast(v16bf, bb);
                acc0[g] = __builtin_amdgcn_wmma_f32_16x16x32_bf16(
                    false, af0, false, bf, (short)0, acc0[g], false, false);
                acc1[g] = __builtin_amdgcn_wmma_f32_16x16x32_bf16(
                    false, af1, false, bf, (short)0, acc1[g], false, false);
            }
        }
        // ---- stage tile t+1 into the other buffer ----
        if (t < 23) {
            __syncthreads();  // WAR: everyone done reading buf^1 (tile t-1)
#pragma unroll
            for (int j = 0; j < 8; ++j)
                ldsB[buf ^ 1][(n0 + j) * LDS_CSTRIDE + kfill] = vnext[j];
            __syncthreads();  // RAW: stores visible before tile t+1 compute
        }
    }

    // ---- fused epilogue: score*fv + x, BN(eval), ReLU ----
#pragma unroll
    for (int mm = 0; mm < 2; ++mm) {
#pragma unroll
        for (int g = 0; g < 4; ++g) {
            const int col = g * 16 + lrow;
#pragma unroll
            for (int i = 0; i < 8; ++i) {
                const int c = m_base + mm * 128 + i + lhalf * 8;
                const float s = scores[b * 256 + c];
                const float inv = gamma[c] * rsqrtf(rvar[c] + 1e-5f);
                const size_t oidx = (((size_t)b * 256 + c) * 64 + h) * 64 + col;
                const float a = mm ? acc1[g][i] : acc0[g][i];
                float v = s * a + x[oidx];
                v = (v - rmean[c]) * inv + beta[c];
                out[oidx] = fmaxf(v, 0.f);
            }
        }
    }
}

// ---------------------------------------------------------------------------
extern "C" void kernel_launch(void* const* d_in, const int* in_sizes, int n_in,
                              void* d_out, int out_size, void* d_ws, size_t ws_size,
                              hipStream_t stream) {
    const float* x     = (const float*)d_in[0];
    const float* wq    = (const float*)d_in[1];
    const float* wk    = (const float*)d_in[2];
    const float* wv    = (const float*)d_in[3];
    const float* gamma = (const float*)d_in[4];
    const float* beta  = (const float*)d_in[5];
    const float* rmean = (const float*)d_in[6];
    const float* rvar  = (const float*)d_in[7];
    float* out = (float*)d_out;

    char* ws = (char*)d_ws;
    unsigned short* xbf    = (unsigned short*)(ws + 0);        // 16M bf16 = 33,554,432 B
    float*          CS     = (float*)(ws + 33554432);          // 16*256*3*64 f32 = 3,145,728 B
    unsigned short* Apk    = (unsigned short*)(ws + 36700160); // 256*2304 bf16 = 1,179,648 B
    float*          wks    = (float*)(ws + 37879808);          // 2304 f32
    float*          ksum   = (float*)(ws + 37889024);          // 16*64 f32
    float*          T      = (float*)(ws + 37893120);          // 16*2304 f32
    float*          scores = (float*)(ws + 38040576);          // 16*256 f32

    k0x_cvt<<<16384, 256, 0, stream>>>(x, xbf);
    k0_packw<<<2304, 256, 0, stream>>>(wv, Apk);
    k1_colsum<<<1024, 256, 0, stream>>>(x, CS);
    k2a_wksum<<<9, 256, 0, stream>>>(wk, wks);
    k2b_ksum<<<4, 256, 0, stream>>>(CS, wks, ksum);
    k3a_T<<<144, 256, 0, stream>>>(CS, ksum, T);
    k3b_scores<<<16, 256, 0, stream>>>(wq, T, scores);

    k4_convv_wmma<<<1024, 256, 0, stream>>>(xbf, x, Apk, scores, gamma, beta,
                                            rmean, rvar, out);
}
